// Attention_49838800503105
// MI455X (gfx1250) — matrix operand order repaired and checked by reading, not compiled, not run on hardware
//
#include <hip/hip_runtime.h>
#include <math.h>

// Problem constants (from reference)
constexpr int Bc = 4, Sc_ = 4096, Dc = 1024, Hc = 16, HDc = 64, LKc = 256;

// ---------- types ----------
typedef __attribute__((ext_vector_type(16))) __bf16 bf16x16;
typedef __attribute__((ext_vector_type(8)))  float  f32x8;
typedef __attribute__((ext_vector_type(4)))  int    v4i;

struct __align__(16) U128 { unsigned int x, y, z, w; };
struct __align__(8)  U64  { unsigned int x, y; };
union FragU { U128 q[2]; bf16x16 v; };

// ---------- async global->LDS (ASYNCcnt path), with safe fallback ----------
#if defined(__has_builtin)
#  if __has_builtin(__builtin_amdgcn_global_load_async_to_lds_b128) && \
      __has_builtin(__builtin_amdgcn_s_wait_asynccnt)
#    define USE_ASYNC_LDS 1
#  endif
#endif
#ifndef USE_ASYNC_LDS
#  define USE_ASYNC_LDS 0
#endif

__device__ __forceinline__ void copy16_to_lds(const unsigned short* gsrc,
                                              unsigned short* ldst) {
#if USE_ASYNC_LDS
    __builtin_amdgcn_global_load_async_to_lds_b128(
        (__attribute__((address_space(1))) v4i*)(void*)gsrc,
        (__attribute__((address_space(3))) v4i*)(void*)ldst, 0, 0);
#else
    *(U128*)ldst = *(const U128*)gsrc;
#endif
}
__device__ __forceinline__ void copy_join() {
#if USE_ASYNC_LDS
    __builtin_amdgcn_s_wait_asynccnt(0);
#endif
}

__device__ __forceinline__ unsigned short f2bf(float f) {
    unsigned int u = __builtin_bit_cast(unsigned int, f);
    u += 0x7FFFu + ((u >> 16) & 1u);        // round-to-nearest-even
    return (unsigned short)(u >> 16);
}

// pack two floats into a dword of two bf16 (lo = a, hi = b), RNE
__device__ __forceinline__ unsigned int pack2bf(float a, float b) {
    unsigned int ua = __builtin_bit_cast(unsigned int, a);
    unsigned int ub = __builtin_bit_cast(unsigned int, b);
    ua += 0x7FFFu + ((ua >> 16) & 1u);
    ub += 0x7FFFu + ((ub >> 16) & 1u);
    return (ua >> 16) | (ub & 0xFFFF0000u);
}

// A fragment: 16x32 (MxK) bf16. ISA layout: lanes 0-15 hold row M=lane,
// K=0..7 & 16..23; lanes 16-31 hold row M=lane-16, K=8..15 & 24..31.
__device__ __forceinline__ bf16x16 load_frag_a(const unsigned short* tile,
                                               int row_base, int stride, int k0, int lane) {
    const int l15 = lane & 15, hi = lane >> 4;
    const unsigned short* p = tile + (size_t)(row_base + l15) * stride + k0 + hi * 8;
    FragU f;
    f.q[0] = *(const U128*)(p);
    f.q[1] = *(const U128*)(p + 16);
    return f.v;
}

// B fragment: 32x16 (KxN) bf16 from a B^T tile (rows = N, cols = K).
// lanes 0-15: col N=lane, K=0..7 & 16..23 ; lanes 16-31: col N=lane-16, K=8..15 & 24..31.
__device__ __forceinline__ bf16x16 load_frag_b(const unsigned short* bt,
                                               int col_base, int stride, int k0, int lane) {
    const int l15 = lane & 15, hi = lane >> 4;
    const unsigned short* p = bt + (size_t)(col_base + l15) * stride + k0 + hi * 8;
    FragU f;
    f.q[0] = *(const U128*)(p);
    f.q[1] = *(const U128*)(p + 16);
    return f.v;
}

__device__ __forceinline__ f32x8 wmma_bf16(bf16x16 a, bf16x16 b, f32x8 c) {
    return __builtin_amdgcn_wmma_f32_16x16x32_bf16(false, a, false, b,
                                                   (short)0, c, false, false);
}

// =====================================================================
// Kernel 1: fused QKV projection.  Y = X @ W + bias, split-heads layout,
// mask folded into K and V.  blockIdx.z selects {Q,K,V}.
// Block tile 128x128, K-step 32, 8 waves (16 rows x 8 N-tiles per wave).
// =====================================================================
__global__ __launch_bounds__(256) void qkv_kernel(
    const float* __restrict__ X, const float* __restrict__ mask,
    const float* __restrict__ Wq, const float* __restrict__ bq,
    const float* __restrict__ Wk, const float* __restrict__ bk,
    const float* __restrict__ Wv, const float* __restrict__ bv,
    unsigned short* __restrict__ Qw, unsigned short* __restrict__ Kw,
    unsigned short* __restrict__ Vw)
{
    __shared__ unsigned short As[128 * 32];
    __shared__ unsigned short Bt[128 * 32];

    const int sel = blockIdx.z;
    const float* W    = (sel == 0) ? Wq : (sel == 1) ? Wk : Wv;
    const float* bias = (sel == 0) ? bq : (sel == 1) ? bk : bv;
    unsigned short* out = (sel == 0) ? Qw : (sel == 1) ? Kw : Vw;

    const int m0 = blockIdx.x * 128;
    const int n0 = blockIdx.y * 128;
    const int t = threadIdx.x, lane = t & 31, wave = t >> 5;
    const int l15 = lane & 15, hi = lane >> 4;

    f32x8 acc[8] = {};

    for (int k0 = 0; k0 < Dc; k0 += 32) {
        // A tile: X[m0..+127][k0..+31] fp32 -> bf16 (1024 float4 chunks,
        // packed pairs -> one 8B DS store per chunk)
        #pragma unroll
        for (int i = 0; i < 4; ++i) {
            int c = t + i * 256;
            int row = c >> 3, kc = (c & 7) << 2;
            float4 v = *(const float4*)&X[(size_t)(m0 + row) * Dc + k0 + kc];
            U64 d; d.x = pack2bf(v.x, v.y); d.y = pack2bf(v.z, v.w);
            *(U64*)&As[row * 32 + kc] = d;
        }
        // B^T tile: for each column n, read 4 consecutive k values (lane-
        // coalesced along n), pack along k -> one 8B DS store.
        #pragma unroll
        for (int i = 0; i < 4; ++i) {
            int c = t + i * 256;                 // 1024 tasks: 128 n x 8 k-groups
            int n = c & 127, kg = (c >> 7) << 2; // kg = 0,4,...,28
            const float* wp = &W[(size_t)(k0 + kg) * (Hc * HDc) + n0 + n];
            float w0 = wp[0];
            float w1 = wp[Hc * HDc];
            float w2 = wp[2 * Hc * HDc];
            float w3 = wp[3 * Hc * HDc];
            U64 d; d.x = pack2bf(w0, w1); d.y = pack2bf(w2, w3);
            *(U64*)&Bt[n * 32 + kg] = d;
        }
        if (k0 + 32 < Dc) {  // keep HBM->L2 pipe primed (global_prefetch_b8)
            __builtin_prefetch(&X[(size_t)(m0 + (t >> 3)) * Dc + (k0 + 32) + ((t & 7) << 2)], 0, 1);
            __builtin_prefetch(&W[(size_t)(k0 + 32 + ((t >> 7) << 2)) * (Hc * HDc) + n0 + (t & 127)], 0, 1);
        }
        __syncthreads();
        bf16x16 afr = load_frag_a(As, wave * 16, 32, 0, lane);
        bf16x16 bfr[8];
        #pragma unroll
        for (int j = 0; j < 8; ++j) bfr[j] = load_frag_b(Bt, j * 16, 32, 0, lane);
        #pragma unroll
        for (int j = 0; j < 8; ++j) acc[j] = wmma_bf16(afr, bfr[j], acc[j]);
        __syncthreads();
    }

    // epilogue: + bias, mask for K/V, store bf16 in (b,h,s,d) layout
    #pragma unroll
    for (int j = 0; j < 8; ++j) {
        #pragma unroll
        for (int r = 0; r < 8; ++r) {
            int m = m0 + wave * 16 + r + 8 * hi;
            int n = n0 + j * 16 + l15;
            int b = m >> 12, s = m & (Sc_ - 1);
            float val = acc[j][r] + bias[n];
            if (sel != 0) val *= mask[(size_t)b * Sc_ + s];
            int h = n >> 6, d = n & 63;
            out[(((size_t)b * Hc + h) * Sc_ + s) * HDc + d] = f2bf(val);
        }
    }
}

// =====================================================================
// Kernel 2: Kp/Vp = E(h) @ (K|V masked).  Per (b,h,{K,V}) block:
// C(256x64), K-loop over S=4096.  8 waves: 32 rows x 4 N-tiles each.
// =====================================================================
__global__ __launch_bounds__(256) void proj_kernel(
    const float* __restrict__ E,
    const unsigned short* __restrict__ Kw, const unsigned short* __restrict__ Vw,
    unsigned short* __restrict__ Kp, unsigned short* __restrict__ Vp)
{
    __shared__ unsigned short As[LKc * 32];   // 256x32
    __shared__ unsigned short Bt[HDc * 32];   // 64x32 (B^T)

    const int bh = blockIdx.x;                // b*H + h
    const int kv = blockIdx.y;                // 0 = K, 1 = V
    const int h  = bh % Hc;
    const float* Eh = E + (size_t)h * LKc * Sc_;
    const unsigned short* Kin = (kv ? Vw : Kw) + (size_t)bh * Sc_ * HDc;
    unsigned short* out = (kv ? Vp : Kp) + (size_t)bh * LKc * HDc;

    const int t = threadIdx.x, lane = t & 31, wave = t >> 5;
    const int l15 = lane & 15, hi = lane >> 4;

    f32x8 acc[2][4] = {};

    for (int s0 = 0; s0 < Sc_; s0 += 32) {
        // A tile: E[0..255][s0..+31] fp32 -> bf16 (2048 float4 chunks)
        #pragma unroll
        for (int i = 0; i < 8; ++i) {
            int c = t + i * 256;
            int row = c >> 3, kc = (c & 7) << 2;
            float4 v = *(const float4*)&Eh[(size_t)row * Sc_ + s0 + kc];
            U64 d; d.x = pack2bf(v.x, v.y); d.y = pack2bf(v.z, v.w);
            *(U64*)&As[row * 32 + kc] = d;
        }
        // B^T tile: Bt[d][s].  Per column d read 4 consecutive s (bf16),
        // pack pairs -> one 8B DS store.  512 tasks: 64 d x 8 s-groups.
        #pragma unroll
        for (int i = 0; i < 2; ++i) {
            int c = t + i * 256;
            int d = c & 63, sg = (c >> 6) << 2;
            const unsigned short* kp = &Kin[(size_t)(s0 + sg) * HDc + d];
            unsigned int v0 = kp[0];
            unsigned int v1 = kp[HDc];
            unsigned int v2 = kp[2 * HDc];
            unsigned int v3 = kp[3 * HDc];
            U64 w; w.x = v0 | (v1 << 16); w.y = v2 | (v3 << 16);
            *(U64*)&Bt[d * 32 + sg] = w;
        }
        if (s0 + 32 < Sc_) {
            __builtin_prefetch(&Eh[(size_t)(t >> 3) * Sc_ + (s0 + 32) + ((t & 7) << 2)], 0, 1);
            __builtin_prefetch(&Kin[(size_t)(s0 + 32 + ((t >> 6) << 2)) * HDc + (t & 63)], 0, 1);
        }
        __syncthreads();
        bf16x16 bfr[4];
        #pragma unroll
        for (int j = 0; j < 4; ++j) bfr[j] = load_frag_b(Bt, j * 16, 32, 0, lane);
        #pragma unroll
        for (int mt = 0; mt < 2; ++mt) {
            bf16x16 afr = load_frag_a(As, wave * 32 + mt * 16, 32, 0, lane);
            #pragma unroll
            for (int j = 0; j < 4; ++j)
                acc[mt][j] = wmma_bf16(afr, bfr[j], acc[mt][j]);
        }
        __syncthreads();
    }

    #pragma unroll
    for (int mt = 0; mt < 2; ++mt)
        #pragma unroll
        for (int j = 0; j < 4; ++j)
            #pragma unroll
            for (int r = 0; r < 8; ++r) {
                int row = wave * 32 + mt * 16 + r + 8 * hi;
                int col = j * 16 + l15;
                out[(size_t)row * HDc + col] = f2bf(acc[mt][j][r]);
            }
}

// =====================================================================
// Kernel 3: attention.  Per (b,h,128-row tile):
//   scores = Q(128x64) . Kp^T  -> softmax (in-register, shfl_xor row
//   reductions matching the C-matrix lane layout) -> P(128x256) . Vp.
// Q and Kp staging are pure copies -> async global->LDS (ASYNCcnt).
// Dynamic LDS: Q 16KB + Kp(B^T) 32KB + P 64KB + Vp^T 32KB = 144KB.
// =====================================================================
__global__ __launch_bounds__(256) void attn_kernel(
    const unsigned short* __restrict__ Qw,
    const unsigned short* __restrict__ Kp, const unsigned short* __restrict__ Vp,
    unsigned short* __restrict__ Xo)
{
    extern __shared__ unsigned short smem[];
    unsigned short* Qa  = smem;                       // 128*64
    unsigned short* KpB = Qa  + 128 * 64;             // 256*64  (already B^T form)
    unsigned short* Pr  = KpB + 256 * 64;             // 128*256 probabilities
    unsigned short* VpT = Pr  + 128 * 256;            // 64*256  (Vp transposed)

    const int s0 = blockIdx.x * 128;
    const int h = blockIdx.y, b = blockIdx.z;
    const int bh = b * Hc + h;
    const unsigned short* Qbh = Qw + (size_t)bh * Sc_ * HDc;
    const unsigned short* Kpb = Kp + (size_t)bh * LKc * HDc;
    const unsigned short* Vpb = Vp + (size_t)bh * LKc * HDc;

    const int t = threadIdx.x, lane = t & 31, wave = t >> 5;
    const int l15 = lane & 15, hi = lane >> 4;

    // stage Q tile (pure copy, 1024 x 16B -> async to LDS)
    #pragma unroll
    for (int i = 0; i < 4; ++i) {
        int c = t + i * 256;
        int row = c >> 3, kc = (c & 7) << 3;
        copy16_to_lds(&Qbh[(size_t)(s0 + row) * HDc + kc], &Qa[row * 64 + kc]);
    }
    // stage Kp (row-major Kp(lk,d) IS the B^T tile for Q.Kp^T)
    #pragma unroll
    for (int i = 0; i < 8; ++i) {
        int c = t + i * 256;
        int row = c >> 3, kc = (c & 7) << 3;
        copy16_to_lds(&Kpb[(size_t)row * HDc + kc], &KpB[row * 64 + kc]);
    }
    // stage Vp transposed: VpT[d][lk].  Per column d read 2 consecutive lk
    // (lane-coalesced along d), pack -> one 4B DS store.  8192 tasks.
    #pragma unroll
    for (int i = 0; i < 32; ++i) {
        int c = t + i * 256;
        int d = c & 63, lk = (c >> 6) << 1;     // lk = 0,2,...,254
        unsigned int v0 = Vpb[(size_t)lk * HDc + d];
        unsigned int v1 = Vpb[(size_t)(lk + 1) * HDc + d];
        *(unsigned int*)&VpT[d * LKc + lk] = v0 | (v1 << 16);
    }
    copy_join();          // s_wait_asynccnt 0 (issuing wave's copies done)
    __syncthreads();

    // GEMM1: 16 rows/wave x 16 N-tiles, K=64 (two k-steps); B frags in
    // groups of 4 so ds_loads batch ahead of the WMMA chain.
    f32x8 acc[16] = {};
    #pragma unroll
    for (int k0 = 0; k0 < HDc; k0 += 32) {
        bf16x16 afr = load_frag_a(Qa, wave * 16, HDc, k0, lane);
        #pragma unroll
        for (int g = 0; g < 4; ++g) {
            bf16x16 bfr[4];
            #pragma unroll
            for (int j = 0; j < 4; ++j)
                bfr[j] = load_frag_b(KpB, (g * 4 + j) * 16, HDc, k0, lane);
            #pragma unroll
            for (int j = 0; j < 4; ++j)
                acc[g * 4 + j] = wmma_bf16(afr, bfr[j], acc[g * 4 + j]);
        }
    }

    // softmax: each (reg r, lane) owns row (wave*16 + r + 8*hi); the 16
    // columns it holds are n = j*16 + l15; lanes {0..15} / {16..31} form
    // independent 16-lane row groups -> shfl_xor masks 1,2,4,8.
    const float scale = 0.125f;   // 1/sqrt(HD)
    #pragma unroll
    for (int r = 0; r < 8; ++r) {
        float e[16];
        float mx = -3.0e38f;
        #pragma unroll
        for (int j = 0; j < 16; ++j) { e[j] = acc[j][r] * scale; mx = fmaxf(mx, e[j]); }
        #pragma unroll
        for (int m = 1; m <= 8; m <<= 1) mx = fmaxf(mx, __shfl_xor(mx, m, 32));
        float sum = 0.f;
        #pragma unroll
        for (int j = 0; j < 16; ++j) { e[j] = __expf(e[j] - mx); sum += e[j]; }
        #pragma unroll
        for (int m = 1; m <= 8; m <<= 1) sum += __shfl_xor(sum, m, 32);
        float inv = 1.0f / sum;
        int row = wave * 16 + r + 8 * hi;
        #pragma unroll
        for (int j = 0; j < 16; ++j)
            Pr[(size_t)row * LKc + j * 16 + l15] = f2bf(e[j] * inv);
    }
    __syncthreads();

    // GEMM2: Xo_tile = P(128x256) . Vp(256x64); 4 N-tiles, 8 k-steps
    f32x8 acc2[4] = {};
    #pragma unroll
    for (int k0 = 0; k0 < LKc; k0 += 32) {
        bf16x16 afr = load_frag_a(Pr, wave * 16, LKc, k0, lane);
        bf16x16 bfr[4];
        #pragma unroll
        for (int j = 0; j < 4; ++j) bfr[j] = load_frag_b(VpT, j * 16, LKc, k0, lane);
        #pragma unroll
        for (int j = 0; j < 4; ++j) acc2[j] = wmma_bf16(afr, bfr[j], acc2[j]);
    }
    // store merged-heads layout: Xo[(b*S + s) * 1024 + h*64 + d]
    #pragma unroll
    for (int j = 0; j < 4; ++j)
        #pragma unroll
        for (int r = 0; r < 8; ++r) {
            int srow = s0 + wave * 16 + r + 8 * hi;
            int d = j * 16 + l15;
            Xo[((size_t)b * Sc_ + srow) * (Hc * HDc) + h * HDc + d] = f2bf(acc2[j][r]);
        }
}

// =====================================================================
// Kernel 4: output projection  out = Xo(bf16) @ Wo + bo  (fp32 out)
// A tile is a pure bf16 copy -> async global->LDS.
// =====================================================================
__global__ __launch_bounds__(256) void outproj_kernel(
    const unsigned short* __restrict__ Xo, const float* __restrict__ Wo,
    const float* __restrict__ bo, float* __restrict__ out)
{
    __shared__ unsigned short As[128 * 32];
    __shared__ unsigned short Bt[128 * 32];

    const int m0 = blockIdx.x * 128;
    const int n0 = blockIdx.y * 128;
    const int t = threadIdx.x, lane = t & 31, wave = t >> 5;
    const int l15 = lane & 15, hi = lane >> 4;

    f32x8 acc[8] = {};

    for (int k0 = 0; k0 < Dc; k0 += 32) {
        // A tile: Xo bf16, pure copy (512 x 16B -> async to LDS)
        #pragma unroll
        for (int i = 0; i < 2; ++i) {
            int c = t + i * 256;
            int row = c >> 2, kc = (c & 3) << 3;
            copy16_to_lds(&Xo[(size_t)(m0 + row) * Dc + k0 + kc], &As[row * 32 + kc]);
        }
        // B^T tile: Wo fp32, column loads + packed 8B DS stores
        #pragma unroll
        for (int i = 0; i < 4; ++i) {
            int c = t + i * 256;
            int n = c & 127, kg = (c >> 7) << 2;
            const float* wp = &Wo[(size_t)(k0 + kg) * Dc + n0 + n];
            float w0 = wp[0];
            float w1 = wp[Dc];
            float w2 = wp[2 * Dc];
            float w3 = wp[3 * Dc];
            U64 d; d.x = pack2bf(w0, w1); d.y = pack2bf(w2, w3);
            *(U64*)&Bt[n * 32 + kg] = d;
        }
        if (k0 + 32 < Dc) {
            __builtin_prefetch(&Xo[(size_t)(m0 + (t >> 2)) * Dc + (k0 + 32) + ((t & 3) << 3)], 0, 1);
            __builtin_prefetch(&Wo[(size_t)(k0 + 32 + ((t >> 7) << 2)) * Dc + n0 + (t & 127)], 0, 1);
        }
        copy_join();      // s_wait_asynccnt 0 before the barrier
        __syncthreads();
        bf16x16 afr = load_frag_a(As, wave * 16, 32, 0, lane);
        bf16x16 bfr[8];
        #pragma unroll
        for (int j = 0; j < 8; ++j) bfr[j] = load_frag_b(Bt, j * 16, 32, 0, lane);
        #pragma unroll
        for (int j = 0; j < 8; ++j) acc[j] = wmma_bf16(afr, bfr[j], acc[j]);
        __syncthreads();
    }

    #pragma unroll
    for (int j = 0; j < 8; ++j)
        #pragma unroll
        for (int r = 0; r < 8; ++r) {
            int m = m0 + wave * 16 + r + 8 * hi;
            int n = n0 + j * 16 + l15;
            out[(size_t)m * Dc + n] = acc[j][r] + bo[n];
        }
}

// =====================================================================
extern "C" void kernel_launch(void* const* d_in, const int* in_sizes, int n_in,
                              void* d_out, int out_size, void* d_ws, size_t ws_size,
                              hipStream_t stream) {
    (void)in_sizes; (void)n_in; (void)out_size; (void)ws_size;
    const float* X    = (const float*)d_in[0];
    const float* mask = (const float*)d_in[1];
    const float* Wq   = (const float*)d_in[2];
    const float* bq   = (const float*)d_in[3];
    const float* Wk   = (const float*)d_in[4];
    const float* bk   = (const float*)d_in[5];
    const float* Wv   = (const float*)d_in[6];
    const float* bv   = (const float*)d_in[7];
    const float* E    = (const float*)d_in[8];
    const float* Wo   = (const float*)d_in[9];
    const float* bo   = (const float*)d_in[10];
    float* out = (float*)d_out;

    // workspace layout (bf16 halves): Q,K,V (b,h,s,d); Kp,Vp (b,h,lk,d); Xo (b,s,h*d)
    const size_t qkvE = (size_t)Bc * Sc_ * Hc * HDc;     // 16.78M elems each
    const size_t pE   = (size_t)Bc * Hc * LKc * HDc;     // 1.05M elems each
    unsigned short* Qw  = (unsigned short*)d_ws;
    unsigned short* Kw  = Qw + qkvE;
    unsigned short* Vw  = Kw + qkvE;
    unsigned short* KpW = Vw + qkvE;
    unsigned short* VpW = KpW + pE;
    unsigned short* XoW = VpW + pE;

    qkv_kernel<<<dim3((Bc * Sc_) / 128, (Hc * HDc) / 128, 3), 256, 0, stream>>>(
        X, mask, Wq, bq, Wk, bk, Wv, bv, Qw, Kw, Vw);

    proj_kernel<<<dim3(Bc * Hc, 2), 256, 0, stream>>>(E, Kw, Vw, KpW, VpW);

    attn_kernel<<<dim3(Sc_ / 128, Hc, Bc), 256, 147456 /* 144KB dyn LDS */, stream>>>(
        Qw, KpW, VpW, XoW);

    outproj_kernel<<<dim3((Bc * Sc_) / 128, Dc / 128), 256, 0, stream>>>(
        XoW, Wo, bo, out);
}
